// PAM_21500606284253
// MI455X (gfx1250) — compile-verified, matrix operand order checked by reading, not run on hardware
//
#include <hip/hip_runtime.h>

typedef __attribute__((ext_vector_type(16))) _Float16 v16h;
typedef __attribute__((ext_vector_type(8)))  _Float16 v8h;
typedef __attribute__((ext_vector_type(8)))  float    v8f;
typedef __attribute__((__vector_size__(16))) int      v4i;

// Problem constants (from reference setup_inputs)
constexpr int Bb = 2;
constexpr int Cc = 64;
constexpr int Hh = 192;
constexpr int Ww = 384;
constexpr int HW = Hh * Ww;        // 73728
constexpr int P  = Bb * HW;        // 147456 pixels

// ---------------------------------------------------------------------------
// Optional gfx1250 async global->LDS copy (ASYNCcnt-tracked). Signature per
// toolchain diagnostic: (v4i* global_src, v4i* lds_dst, imm offset, imm cpol).
// Falls back to a plain vector LDS copy if the builtin is not declared.
// ---------------------------------------------------------------------------
#if __has_builtin(__builtin_amdgcn_global_load_async_to_lds_b128)
#define ASYNC_LDS_CP16(gsrc, ldst)                                             \
    __builtin_amdgcn_global_load_async_to_lds_b128((v4i*)(gsrc), (v4i*)(ldst), \
                                                   0, 0)
#define ASYNC_LDS_WAIT() asm volatile("s_wait_asynccnt 0" ::: "memory")
#else
#define ASYNC_LDS_CP16(gsrc, ldst) (*(v8h*)(ldst) = *(const v8h*)(gsrc))
#define ASYNC_LDS_WAIT()
#endif

// ---------------------------------------------------------------------------
// WMMA f16 16x16x32 fragment layout (A and B identical):
//   lane l holds row/col (l&15); elements 0..7 = K hb..hb+7,
//   elements 8..15 = K hb+16..hb+23, where hb = 8*(l>>4).
// With K-contiguous storage a fragment is TWO 16-byte vector loads.
// ---------------------------------------------------------------------------
__device__ __forceinline__ v16h frag_row(const _Float16* rowk, int hb) {
    v8h lo = *(const v8h*)(rowk + hb);
    v8h hi = *(const v8h*)(rowk + 16 + hb);
    return __builtin_shufflevector(lo, hi, 0, 1, 2, 3, 4, 5, 6, 7,
                                           8, 9, 10, 11, 12, 13, 14, 15);
}

__device__ __forceinline__ v8f wmma_f16(v16h a, v16h b, v8f c) {
    return __builtin_amdgcn_wmma_f32_16x16x32_f16(false, a, false, b, (short)0, c,
                                                  false, false);
}

// ---------------------------------------------------------------------------
// Packing: fp32 NCHW -> f16 pixel-major [p][64]; weights K-contiguous rows.
// ---------------------------------------------------------------------------
__global__ void pack_act(const float* __restrict__ x, _Float16* __restrict__ xh,
                         int total) {
    int i = blockIdx.x * blockDim.x + threadIdx.x;
    if (i >= total) return;
    int p  = i >> 6, c = i & 63;
    int b  = p / HW;
    int hw = p - b * HW;
    xh[i] = (_Float16)x[(b * Cc + c) * HW + hw];
}

__global__ void pack_weights(const float* __restrict__ w1, const float* __restrict__ w2,
                             const float* __restrict__ b1, const float* __restrict__ b2,
                             const float* __restrict__ b3, const float* __restrict__ fw,
                             _Float16* __restrict__ w1h, _Float16* __restrict__ w2h,
                             _Float16* __restrict__ b1h, _Float16* __restrict__ b2h,
                             _Float16* __restrict__ b3h, _Float16* __restrict__ fwh) {
    int i = blockIdx.x * blockDim.x + threadIdx.x;
    if (i < 9 * 64 * 64) {  // OIHW -> [co][t*64+ci]  (K = tap-major, ci contiguous)
        int co = i / 576, rr = i % 576, t = rr >> 6, ci = rr & 63;
        int src = (co * 64 + ci) * 9 + t;
        w1h[i] = (_Float16)w1[src];
        w2h[i] = (_Float16)w2[src];
    }
    if (i < 4096) {         // 1x1 weights stay [co][ci] (ci contiguous)
        b1h[i] = (_Float16)b1[i];
        b2h[i] = (_Float16)b2[i];
        b3h[i] = (_Float16)b3[i];
    }
    if (i < 64 * 128) {     // fusion [co][129] -> [co][128]
        int co = i >> 7, c = i & 127;
        fwh[i] = (_Float16)fw[co * 129 + c];
    }
}

__global__ void zero_f32(float* __restrict__ p, int n) {
    int i = blockIdx.x * blockDim.x + threadIdx.x;
    if (i < n) p[i] = 0.0f;
}

__global__ void mask_kernel(const float* __restrict__ cs, float* __restrict__ V, int n) {
    int i = blockIdx.x * blockDim.x + threadIdx.x;
    if (i < n) V[i] = cs[i] > 0.1f ? 1.0f : 0.0f;
}

// ---------------------------------------------------------------------------
// 3x3 conv as implicit GEMM. Block = 128 thr (4 wave32), grid (W/32, H, B).
// Wave w: co-tile [16w,16w+16) x 32 width positions -> each weight fragment
// feeds TWO WMMAs. Halo tile staged via async global->LDS b128 copies.
// ---------------------------------------------------------------------------
template <bool LRELU, bool RESID>
__global__ __launch_bounds__(128)
void conv3x3_wmma(const _Float16* __restrict__ in,    // [P][64]
                  const _Float16* __restrict__ wt,    // [64 co][576]
                  const _Float16* __restrict__ resid, // [P][64] or null
                  _Float16* __restrict__ out) {       // [P][64]
    __shared__ _Float16 sX[34][3][64];  // [dx][dy][ci], ci contiguous
    const int x0 = blockIdx.x * 32;
    const int y  = blockIdx.y;
    const int b  = blockIdx.z;
    const int tid = threadIdx.x, lane = tid & 31, wv = tid >> 5;
    const int hb = (lane >> 4) << 3, n = lane & 15;
    const int rowbase = (b * Hh + y) * Ww;

    // stage halo: 102 pixel-rows of 64 f16 (128B) = 816 16B chunks
    for (int idx = tid; idx < 34 * 3 * 8; idx += 128) {
        int c8 = (idx & 7) << 3;
        int r = idx >> 3;
        int dx = r / 3, dy = r % 3;
        int yy = y + dy - 1, xx = x0 + dx - 1;
        _Float16* dst = &sX[dx][dy][c8];
        if (yy >= 0 && yy < Hh && xx >= 0 && xx < Ww) {
            ASYNC_LDS_CP16(in + ((b * Hh + yy) * Ww + xx) * 64 + c8, dst);
        } else {
            v8h z = {};
            *(v8h*)dst = z;
        }
    }
    ASYNC_LDS_WAIT();
    __syncthreads();

    const int co0 = wv * 16;
    const _Float16* wrow = wt + (co0 + n) * 576;  // A row (m = n)
    __builtin_prefetch(wrow, 0, 0);
    v8f acc[2] = {{}, {}};
    for (int t = 0; t < 9; ++t) {
        const int ky = t / 3, kx = t % 3;
#pragma unroll
        for (int cb = 0; cb < 2; ++cb) {
            v16h a  = frag_row(wrow + t * 64 + cb * 32, hb);
            v16h b0 = frag_row(&sX[kx + n][ky][cb * 32], hb);
            v16h b1 = frag_row(&sX[kx + 16 + n][ky][cb * 32], hb);
            acc[0] = wmma_f16(a, b0, acc[0]);
            acc[1] = wmma_f16(a, b1, acc[1]);
        }
    }
    // epilogue: per pixel, 8 contiguous channels -> single b128 store
#pragma unroll
    for (int px = 0; px < 2; ++px) {
        const int pix = rowbase + x0 + px * 16 + n;
        v8f v = acc[px];
        if constexpr (RESID) {
            v += __builtin_convertvector(
                *(const v8h*)(resid + pix * 64 + co0 + hb), v8f);
        }
        if constexpr (LRELU) {
#pragma unroll
            for (int r = 0; r < 8; ++r) v[r] = v[r] > 0.0f ? v[r] : 0.1f * v[r];
        }
        *(v8h*)(out + pix * 64 + co0 + hb) = __builtin_convertvector(v, v8h);
    }
}

// ---------------------------------------------------------------------------
// 1x1 conv (channel GEMM). One wave per 16-pixel tile computing ALL four
// co-tiles: B fragments loaded once, reused 4x. No LDS needed.
// OUT_CHMAJOR=true stores [64][P] (needed for Vr, consumed j-contiguous).
// ---------------------------------------------------------------------------
template <bool OUT_CHMAJOR>
__global__ __launch_bounds__(128)
void proj1x1_wmma(const _Float16* __restrict__ in,  // [P][64]
                  const _Float16* __restrict__ wt,  // [64 co][64 ci]
                  const float* __restrict__ bias,   // [64]
                  _Float16* __restrict__ out) {
    const int tid = threadIdx.x, lane = tid & 31, wv = tid >> 5;
    const int hb = (lane >> 4) << 3, n = lane & 15;
    const int p0 = blockIdx.x * 64 + wv * 16;
    const _Float16* irow = in + (p0 + n) * 64;
    const v16h b0 = frag_row(irow, hb);
    const v16h b1 = frag_row(irow + 32, hb);
#pragma unroll
    for (int ct = 0; ct < 4; ++ct) {
        const _Float16* wrow = wt + (ct * 16 + n) * 64;
        v8f acc = {};
        acc = wmma_f16(frag_row(wrow, hb), b0, acc);
        acc = wmma_f16(frag_row(wrow + 32, hb), b1, acc);
        if constexpr (OUT_CHMAJOR) {
#pragma unroll
            for (int r = 0; r < 8; ++r)
                out[(ct * 16 + hb + r) * P + p0 + n] =
                    (_Float16)(acc[r] + bias[ct * 16 + hb + r]);
        } else {
            v8h o;
#pragma unroll
            for (int r = 0; r < 8; ++r)
                o[r] = (_Float16)(acc[r] + bias[ct * 16 + hb + r]);
            *(v8h*)(out + (p0 + n) * 64 + ct * 16 + hb) = o;
        }
    }
}

// ---------------------------------------------------------------------------
// Per-row parallax attention: one block per (b, h, 16-row i-block).
// Q fragments hoisted (reused 6x); scores -> LDS; parallel softmax;
// then column-sum atomics (validity) or P @ Vr via WMMA (warp).
// ---------------------------------------------------------------------------
template <bool WRITE_BUF, bool WRITE_COLSUM>
__global__ __launch_bounds__(128)
void attn_row_wmma(const _Float16* __restrict__ q,   // [P][64]
                   const _Float16* __restrict__ s,   // [P][64]
                   const _Float16* __restrict__ v,   // [64][P] ch-major or null
                   float* __restrict__ colsum,       // [P] or null
                   _Float16* __restrict__ buf) {     // [P][64] or null
    __shared__ float    sScores[16][Ww];
    __shared__ _Float16 sProbs[16][Ww];
    __shared__ float    sRed[16][8];
    __shared__ float    sVal[16];
    const int i0 = blockIdx.x * 16;
    const int h = blockIdx.y, b = blockIdx.z;
    const int base = (b * Hh + h) * Ww;
    const int tid = threadIdx.x, lane = tid & 31, wv = tid >> 5;
    const int hb = (lane >> 4) << 3, n = lane & 15;

    // A fragments (m = i row): load once, reuse for all j-tiles
    const _Float16* qrow = q + (base + i0 + n) * 64;
    const v16h aq0 = frag_row(qrow, hb);
    const v16h aq1 = frag_row(qrow + 32, hb);

    for (int jt = wv; jt < Ww / 16; jt += 4) {
        const _Float16* srow = s + (base + jt * 16 + n) * 64;
        v8f acc = {};
        acc = wmma_f16(aq0, frag_row(srow, hb), acc);
        acc = wmma_f16(aq1, frag_row(srow + 32, hb), acc);
#pragma unroll
        for (int r = 0; r < 8; ++r) sScores[hb + r][jt * 16 + n] = acc[r];
    }
    __syncthreads();

    // Parallel softmax: 8 threads per row, 48 columns each
    const int row = tid >> 3, sub = tid & 7;
    const int j0 = sub * 48;
    float mx = -1e30f;
    for (int j = 0; j < 48; ++j) mx = fmaxf(mx, sScores[row][j0 + j]);
    sRed[row][sub] = mx;
    __syncthreads();
    if (sub == 0) {
        float m = sRed[row][0];
        for (int k = 1; k < 8; ++k) m = fmaxf(m, sRed[row][k]);
        sVal[row] = m;
    }
    __syncthreads();
    const float rmax = sVal[row];
    float sum = 0.0f;
    for (int j = 0; j < 48; ++j) sum += __expf(sScores[row][j0 + j] - rmax);
    sRed[row][sub] = sum;
    __syncthreads();
    if (sub == 0) {
        float t = 0.0f;
        for (int k = 0; k < 8; ++k) t += sRed[row][k];
        sVal[row] = 1.0f / t;
    }
    __syncthreads();
    const float inv = sVal[row];
    for (int j = 0; j < 48; ++j)
        sProbs[row][j0 + j] = (_Float16)(__expf(sScores[row][j0 + j] - rmax) * inv);
    __syncthreads();

    if constexpr (WRITE_COLSUM) {  // partial column sums over this i-block
        for (int j = tid; j < Ww; j += 128) {
            float sm = 0.0f;
#pragma unroll
            for (int i = 0; i < 16; ++i) sm += (float)sProbs[i][j];
            atomicAdd(&colsum[base + j], sm);
        }
    }

    if constexpr (WRITE_BUF) {  // buf[16 i][64 c] = P[16x384] @ Vr^T
        const int co0 = wv * 16;
        v8f acc = {};
        for (int kt = 0; kt < Ww / 32; ++kt) {
            v16h a  = frag_row(&sProbs[n][kt * 32], hb);                // A(m=i,k=j)
            v16h bf = frag_row(v + (co0 + n) * P + base + kt * 32, hb); // B(k=j,n=c)
            acc = wmma_f16(a, bf, acc);
        }
#pragma unroll
        for (int r = 0; r < 8; ++r)
            buf[(base + i0 + hb + r) * 64 + co0 + n] = (_Float16)acc[r];
    }
}

// ---------------------------------------------------------------------------
// Fusion: out[co][p] = fusW[co][0:64].buf + fusW[co][64:128].x_left
//                    + fusW[co][128]*V[p] + fus_b[co]   (fp32 NCHW output)
// One wave per 16-pixel tile, all four co-tiles: B fragments reused 4x.
// ---------------------------------------------------------------------------
__global__ __launch_bounds__(128)
void fuse_wmma(const _Float16* __restrict__ buf,  // [P][64]
               const _Float16* __restrict__ xl,   // [P][64]
               const float* __restrict__ vmask,   // [P]
               const _Float16* __restrict__ fwh,  // [64 co][128]
               const float* __restrict__ fw,      // [64][129] fp32 (col 128)
               const float* __restrict__ fb,      // [64]
               float* __restrict__ out) {         // [b][64][h][w]
    const int tid = threadIdx.x, lane = tid & 31, wv = tid >> 5;
    const int hb = (lane >> 4) << 3, n = lane & 15;
    const int p0 = blockIdx.x * 64 + wv * 16;
    const _Float16* brow = buf + (p0 + n) * 64;
    const _Float16* xrow = xl + (p0 + n) * 64;
    const v16h fb0 = frag_row(brow, hb);
    const v16h fb1 = frag_row(brow + 32, hb);
    const v16h fx0 = frag_row(xrow, hb);
    const v16h fx1 = frag_row(xrow + 32, hb);
    const int p = p0 + n;
    const float vm = vmask[p];
    const int bb = p / HW, hw = p - bb * HW;
#pragma unroll
    for (int ct = 0; ct < 4; ++ct) {
        const _Float16* wrow = fwh + (ct * 16 + n) * 128;
        v8f acc = {};
        acc = wmma_f16(frag_row(wrow,      hb), fb0, acc);
        acc = wmma_f16(frag_row(wrow + 32, hb), fb1, acc);
        acc = wmma_f16(frag_row(wrow + 64, hb), fx0, acc);
        acc = wmma_f16(frag_row(wrow + 96, hb), fx1, acc);
#pragma unroll
        for (int r = 0; r < 8; ++r) {
            int co = ct * 16 + hb + r;
            out[(bb * Cc + co) * HW + hw] = acc[r] + fw[co * 129 + 128] * vm + fb[co];
        }
    }
}

// ---------------------------------------------------------------------------
// Host orchestration. Workspace (f16 activation buffer = 18.87 MB):
//   [0] xl  [1] xr  [2] T (conv tmp -> qr)  [3] bl (-> vr, ch-major)
//   [4] br  [5] ql  [6] sr                  [7] sl (-> buf)
// + packed weights + colsum + vmask  (total ~152.4 MB)
// ---------------------------------------------------------------------------
extern "C" void kernel_launch(void* const* d_in, const int* in_sizes, int n_in,
                              void* d_out, int out_size, void* d_ws, size_t ws_size,
                              hipStream_t stream) {
    const float* x_left  = (const float*)d_in[0];
    const float* x_right = (const float*)d_in[1];
    const float* rb_w1   = (const float*)d_in[2];
    const float* rb_w2   = (const float*)d_in[3];
    const float* b1_w    = (const float*)d_in[4];
    const float* b1_b    = (const float*)d_in[5];
    const float* b2_w    = (const float*)d_in[6];
    const float* b2_b    = (const float*)d_in[7];
    const float* b3_w    = (const float*)d_in[8];
    const float* b3_b    = (const float*)d_in[9];
    const float* fus_w   = (const float*)d_in[10];
    const float* fus_b   = (const float*)d_in[11];
    float* out = (float*)d_out;

    char* ws = (char*)d_ws;
    const size_t ACT = (size_t)Cc * P * sizeof(_Float16);
    _Float16* xl_h = (_Float16*)(ws + 0 * ACT);
    _Float16* xr_h = (_Float16*)(ws + 1 * ACT);
    _Float16* T_h  = (_Float16*)(ws + 2 * ACT);  // conv tmp, then qr
    _Float16* bl_h = (_Float16*)(ws + 3 * ACT);  // bl, then vr (ch-major)
    _Float16* br_h = (_Float16*)(ws + 4 * ACT);
    _Float16* ql_h = (_Float16*)(ws + 5 * ACT);
    _Float16* sr_h = (_Float16*)(ws + 6 * ACT);
    _Float16* sl_h = (_Float16*)(ws + 7 * ACT);  // sl, then buf
    char* sm = ws + 8 * ACT;
    _Float16* w1_h = (_Float16*)sm; sm += 9 * 64 * 64 * 2;
    _Float16* w2_h = (_Float16*)sm; sm += 9 * 64 * 64 * 2;
    _Float16* b1_h = (_Float16*)sm; sm += 64 * 64 * 2;
    _Float16* b2_h = (_Float16*)sm; sm += 64 * 64 * 2;
    _Float16* b3_h = (_Float16*)sm; sm += 64 * 64 * 2;
    _Float16* fw_h = (_Float16*)sm; sm += 64 * 128 * 2;
    float* colsum  = (float*)sm;    sm += (size_t)P * 4;
    float* vmask   = (float*)sm;

    const dim3 blk(128);
    const int total = Cc * P;
    pack_act<<<(total + 255) / 256, 256, 0, stream>>>(x_left, xl_h, total);
    pack_act<<<(total + 255) / 256, 256, 0, stream>>>(x_right, xr_h, total);
    pack_weights<<<(9 * 64 * 64 + 255) / 256, 256, 0, stream>>>(
        rb_w1, rb_w2, b1_w, b2_w, b3_w, fus_w, w1_h, w2_h, b1_h, b2_h, b3_h, fw_h);

    // Shared ResB on both views
    const dim3 cgrid(Ww / 32, Hh, Bb);
    conv3x3_wmma<true,  false><<<cgrid, blk, 0, stream>>>(xl_h, w1_h, nullptr, T_h);
    conv3x3_wmma<false, true ><<<cgrid, blk, 0, stream>>>(T_h,  w2_h, xl_h,    bl_h);
    conv3x3_wmma<true,  false><<<cgrid, blk, 0, stream>>>(xr_h, w1_h, nullptr, T_h);
    conv3x3_wmma<false, true ><<<cgrid, blk, 0, stream>>>(T_h,  w2_h, xr_h,    br_h);

    // 1x1 projections (bl consumed before vr overwrites it)
    const dim3 pgrid(P / 64);
    proj1x1_wmma<false><<<pgrid, blk, 0, stream>>>(bl_h, b1_h, b1_b, ql_h);
    proj1x1_wmma<false><<<pgrid, blk, 0, stream>>>(bl_h, b2_h, b2_b, sl_h);
    proj1x1_wmma<false><<<pgrid, blk, 0, stream>>>(br_h, b1_h, b1_b, T_h);   // qr
    proj1x1_wmma<false><<<pgrid, blk, 0, stream>>>(br_h, b2_h, b2_b, sr_h);
    proj1x1_wmma<true ><<<pgrid, blk, 0, stream>>>(xr_h, b3_h, b3_b, bl_h);  // vr

    // Validity mask from l->r attention column sums
    zero_f32<<<(P + 255) / 256, 256, 0, stream>>>(colsum, P);
    const dim3 agrid(Ww / 16, Hh, Bb);
    attn_row_wmma<false, true><<<agrid, blk, 0, stream>>>(
        T_h /*qr*/, sl_h /*sl*/, nullptr, colsum, nullptr);
    mask_kernel<<<(P + 255) / 256, 256, 0, stream>>>(colsum, vmask, P);

    // r->l attention warps Vr into buf (reuses sl_h storage)
    attn_row_wmma<true, false><<<agrid, blk, 0, stream>>>(
        ql_h, sr_h, bl_h /*vr*/, nullptr, sl_h /*buf*/);

    // Fusion 1x1 over [buf, x_left, V]
    fuse_wmma<<<pgrid, blk, 0, stream>>>(sl_h /*buf*/, xl_h, vmask, fw_h,
                                         fus_w, fus_b, out);
}
